// SVDHead_10196252361152
// MI455X (gfx1250) — matrix operand order repaired and checked by reading, not compiled, not run on hardware
//
#include <hip/hip_runtime.h>
#include <cstdint>
#include <cstddef>

// Problem constants (match reference)
#define BB 32
#define DD 512
#define NN 2048
#define MM 2048

typedef __attribute__((ext_vector_type(16))) __bf16 v16bf;
typedef __attribute__((ext_vector_type(8)))  float  v8f;

// ---------- bf16 helpers (round-to-nearest-even, hi/lo split) ----------
__device__ __forceinline__ uint16_t f2bf(float f) {
    uint32_t u = __float_as_uint(f);
    uint32_t r = u + 0x7FFFu + ((u >> 16) & 1u);
    return (uint16_t)(r >> 16);
}
__device__ __forceinline__ float bf2f(uint16_t h) {
    return __uint_as_float(((uint32_t)h) << 16);
}

union FragU {
    uint4 q[2];
    v16bf v;
};

// A-fragment (16x32 bf16): lane l holds row (l&15); elems 0..7 = K kb..kb+7,
// elems 8..15 = K 16+kb..16+kb+7, kb = (l>=16)*8.  Two aligned 16B LDS reads.
__device__ __forceinline__ v16bf load_frag_a(const uint16_t* row, int kb) {
    FragU f;
    f.q[0] = *(const uint4*)(row + kb);
    f.q[1] = *(const uint4*)(row + 16 + kb);
    return f.v;
}
// B-fragment (32x16 bf16): lane l holds col (l&15); elems 0..7 = K kb2..kb2+7,
// elems 8..15 = K 16+kb2..16+kb2+7, kb2 = (l>=16)*8.
__device__ __forceinline__ v16bf load_frag_b(const uint16_t* row, int kb2) {
    FragU f;
    f.q[0] = *(const uint4*)(row + kb2);
    f.q[1] = *(const uint4*)(row + 16 + kb2);
    return f.v;
}

__device__ __forceinline__ void pack_store8(uint16_t* dst, const uint16_t* v) {
    uint4 q;
    q.x = (uint32_t)v[0] | ((uint32_t)v[1] << 16);
    q.y = (uint32_t)v[2] | ((uint32_t)v[3] << 16);
    q.z = (uint32_t)v[4] | ((uint32_t)v[5] << 16);
    q.w = (uint32_t)v[6] | ((uint32_t)v[7] << 16);
    *(uint4*)dst = q;
}

// =====================================================================
// Kernel 1: scores[b,n,m] = (src_emb[b,:,n] . tgt_emb[b,:,m]) / sqrt(512)
// bf16 hi/lo split GEMM on v_wmma_f32_16x16x32_bf16.
// Block tile: 64(n) x 128(m), 8 waves, each wave 32x32 (2x2 WMMA tiles).
// =====================================================================
#define AST 40   // padded LDS row stride (bf16 elems), 80B = multiple of 16B
#define BST 40

__global__ __launch_bounds__(256)
void svdhead_scores_wmma(const float* __restrict__ srcE,
                         const float* __restrict__ tgtE,
                         float* __restrict__ scores) {
    __shared__ alignas(16) uint16_t Ahi[64 * AST];
    __shared__ alignas(16) uint16_t Alo[64 * AST];
    __shared__ alignas(16) uint16_t Bhi[128 * BST];
    __shared__ alignas(16) uint16_t Blo[128 * BST];

    const int b  = blockIdx.z;
    const int n0 = blockIdx.y * 64;
    const int m0 = blockIdx.x * 128;
    const int tid = threadIdx.x;

    const float* Ag = srcE + (size_t)b * DD * NN;  // [d][n]
    const float* Bg = tgtE + (size_t)b * DD * MM;  // [d][m]

    // wave coordinates
    const int w     = tid >> 5;
    const int lane  = tid & 31;
    const int waveN = w >> 2;        // 0..1  -> 32 rows of n
    const int waveM = w & 3;         // 0..3  -> 32 cols of m
    const int r     = lane & 15;
    const int kbA   = (lane >> 4) * 8;
    const int kbB   = (lane >> 4) * 8;

    // loader coordinates (coalesced global, contiguous 16B LDS stores)
    const int an  = tid & 63;        // A: n index
    const int akc = tid >> 6;        // A: k chunk 0..3 (8 k's each)
    const int bm  = tid & 127;       // B: m index
    const int bkc = tid >> 7;        // B: k chunk 0..1 (16 k's each)

    v8f acc[2][2];
    acc[0][0] = {}; acc[0][1] = {}; acc[1][0] = {}; acc[1][1] = {};

    for (int k0 = 0; k0 < DD; k0 += 32) {
        // ---- stage A tile: 32(k) x 64(n) -> LDS [n][k] hi/lo ----
        {
            uint16_t hi[8], lo[8];
            #pragma unroll
            for (int i = 0; i < 8; ++i) {
                float x = Ag[(size_t)(k0 + akc * 8 + i) * NN + (n0 + an)];
                uint16_t h = f2bf(x);
                hi[i] = h;
                lo[i] = f2bf(x - bf2f(h));
            }
            pack_store8(&Ahi[an * AST + akc * 8], hi);
            pack_store8(&Alo[an * AST + akc * 8], lo);
        }
        // ---- stage B tile: 32(k) x 128(m) -> LDS [m][k] hi/lo ----
        {
            uint16_t hi[16], lo[16];
            #pragma unroll
            for (int i = 0; i < 16; ++i) {
                float x = Bg[(size_t)(k0 + bkc * 16 + i) * MM + (m0 + bm)];
                uint16_t h = f2bf(x);
                hi[i] = h;
                lo[i] = f2bf(x - bf2f(h));
            }
            pack_store8(&Bhi[bm * BST + bkc * 16], hi);
            pack_store8(&Bhi[bm * BST + bkc * 16 + 8], hi + 8);
            pack_store8(&Blo[bm * BST + bkc * 16], lo);
            pack_store8(&Blo[bm * BST + bkc * 16 + 8], lo + 8);
        }
        __syncthreads();

        // ---- fragments + 12 WMMAs (hi*hi + hi*lo + lo*hi) ----
        v16bf aH[2], aL[2], bH[2], bL[2];
        #pragma unroll
        for (int tn = 0; tn < 2; ++tn) {
            const uint16_t* rowH = &Ahi[(waveN * 32 + tn * 16 + r) * AST];
            const uint16_t* rowL = &Alo[(waveN * 32 + tn * 16 + r) * AST];
            aH[tn] = load_frag_a(rowH, kbA);
            aL[tn] = load_frag_a(rowL, kbA);
        }
        #pragma unroll
        for (int tm = 0; tm < 2; ++tm) {
            const uint16_t* rowH = &Bhi[(waveM * 32 + tm * 16 + r) * BST];
            const uint16_t* rowL = &Blo[(waveM * 32 + tm * 16 + r) * BST];
            bH[tm] = load_frag_b(rowH, kbB);
            bL[tm] = load_frag_b(rowL, kbB);
        }
        #pragma unroll
        for (int tn = 0; tn < 2; ++tn) {
            #pragma unroll
            for (int tm = 0; tm < 2; ++tm) {
                acc[tn][tm] = __builtin_amdgcn_wmma_f32_16x16x32_bf16(
                    false, aH[tn], false, bH[tm], (short)0, acc[tn][tm], false, false);
                acc[tn][tm] = __builtin_amdgcn_wmma_f32_16x16x32_bf16(
                    false, aH[tn], false, bL[tm], (short)0, acc[tn][tm], false, false);
                acc[tn][tm] = __builtin_amdgcn_wmma_f32_16x16x32_bf16(
                    false, aL[tn], false, bH[tm], (short)0, acc[tn][tm], false, false);
            }
        }
        __syncthreads();
    }

    // ---- epilogue: scale by 1/sqrt(512), write C ----
    const float scale = 0.044194173824159216f;  // 1/sqrt(512)
    #pragma unroll
    for (int tn = 0; tn < 2; ++tn) {
        #pragma unroll
        for (int tm = 0; tm < 2; ++tm) {
            const int gm = m0 + waveM * 32 + tm * 16 + (lane & 15);
            const int nb = n0 + waveN * 32 + tn * 16 + (lane >> 4) * 8;
            #pragma unroll
            for (int e = 0; e < 8; ++e) {
                const int gn = nb + e;
                scores[((size_t)b * NN + gn) * MM + gm] = acc[tn][tm][e] * scale;
            }
        }
    }
}

// =====================================================================
// Kernel 2: fused softmax(row) + src_corr[b,d,n] = sum_m attn * tgt[b,d,m]
// One block per (b,n) row; scores cached in registers (single HBM read).
// =====================================================================
__global__ __launch_bounds__(256)
void svdhead_softmax_corr(const float* __restrict__ scores,
                          const float* __restrict__ tgt,
                          float* __restrict__ corr) {
    const int row = blockIdx.x;          // b*N + n
    const int b = row >> 11;             // /2048
    const int n = row & 2047;
    const int tid = threadIdx.x;
    const int base = tid * 8;

    const float* srow = scores + (size_t)row * MM;
    const float* tb   = tgt + (size_t)b * 3 * MM;

    float sv[8];
    #pragma unroll
    for (int i = 0; i < 8; ++i) sv[i] = srow[base + i];

    float lmax = -3.4e38f;
    #pragma unroll
    for (int i = 0; i < 8; ++i) lmax = fmaxf(lmax, sv[i]);

    __shared__ float red[256];
    red[tid] = lmax;
    __syncthreads();
    for (int s = 128; s > 0; s >>= 1) {
        if (tid < s) red[tid] = fmaxf(red[tid], red[tid + s]);
        __syncthreads();
    }
    const float mx = red[0];
    __syncthreads();

    float lsum = 0.f, l0 = 0.f, l1 = 0.f, l2 = 0.f;
    #pragma unroll
    for (int i = 0; i < 8; ++i) {
        float e = expf(sv[i] - mx);
        lsum += e;
        l0 += e * tb[base + i];
        l1 += e * tb[MM + base + i];
        l2 += e * tb[2 * MM + base + i];
    }

    __shared__ float rs[256], r0[256], r1[256], r2[256];
    rs[tid] = lsum; r0[tid] = l0; r1[tid] = l1; r2[tid] = l2;
    __syncthreads();
    for (int s = 128; s > 0; s >>= 1) {
        if (tid < s) {
            rs[tid] += rs[tid + s];
            r0[tid] += r0[tid + s];
            r1[tid] += r1[tid + s];
            r2[tid] += r2[tid + s];
        }
        __syncthreads();
    }
    if (tid == 0) {
        const float inv = 1.0f / rs[0];
        corr[((size_t)b * 3 + 0) * NN + n] = r0[0] * inv;
        corr[((size_t)b * 3 + 1) * NN + n] = r1[0] * inv;
        corr[((size_t)b * 3 + 2) * NN + n] = r2[0] * inv;
    }
}

// =====================================================================
// Kernel 3: per-batch Kabsch: means, H, 3x3 SVD (Jacobi on H^T H),
// reflection fix, R and t.  One block per batch.
// =====================================================================
__global__ __launch_bounds__(256)
void svdhead_kabsch(const float* __restrict__ src,
                    const float* __restrict__ corr,
                    float* __restrict__ Rout,
                    float* __restrict__ tout) {
    const int b = blockIdx.x;
    const int tid = threadIdx.x;
    const float* S = src  + (size_t)b * 3 * NN;
    const float* C = corr + (size_t)b * 3 * NN;

    float acc[15];
    #pragma unroll
    for (int k = 0; k < 15; ++k) acc[k] = 0.f;

    for (int n = tid; n < NN; n += 256) {
        float s0 = S[n], s1 = S[NN + n], s2 = S[2 * NN + n];
        float c0 = C[n], c1 = C[NN + n], c2 = C[2 * NN + n];
        acc[0] += s0; acc[1] += s1; acc[2] += s2;
        acc[3] += c0; acc[4] += c1; acc[5] += c2;
        acc[6]  += s0 * c0; acc[7]  += s0 * c1; acc[8]  += s0 * c2;
        acc[9]  += s1 * c0; acc[10] += s1 * c1; acc[11] += s1 * c2;
        acc[12] += s2 * c0; acc[13] += s2 * c1; acc[14] += s2 * c2;
    }

    __shared__ float red[256];
    __shared__ float tot[15];
    for (int k = 0; k < 15; ++k) {
        red[tid] = acc[k];
        __syncthreads();
        for (int s = 128; s > 0; s >>= 1) {
            if (tid < s) red[tid] += red[tid + s];
            __syncthreads();
        }
        if (tid == 0) tot[k] = red[0];
        __syncthreads();
    }
    if (tid != 0) return;

    const float invN = 1.0f / (float)NN;
    float sm[3] = {tot[0] * invN, tot[1] * invN, tot[2] * invN};
    float cm[3] = {tot[3] * invN, tot[4] * invN, tot[5] * invN};
    float Hm[3][3];
    for (int i = 0; i < 3; ++i)
        for (int j = 0; j < 3; ++j)
            Hm[i][j] = tot[6 + i * 3 + j] - (float)NN * sm[i] * cm[j];

    // A = H^T H ; Jacobi eigen-decomposition -> V (right singular vectors)
    float Am[3][3];
    for (int i = 0; i < 3; ++i)
        for (int j = 0; j < 3; ++j)
            Am[i][j] = Hm[0][i] * Hm[0][j] + Hm[1][i] * Hm[1][j] + Hm[2][i] * Hm[2][j];

    float V[3][3] = {{1.f, 0.f, 0.f}, {0.f, 1.f, 0.f}, {0.f, 0.f, 1.f}};
    const int PP[3] = {0, 0, 1}, QQ[3] = {1, 2, 2};
    for (int sweep = 0; sweep < 12; ++sweep) {
        for (int pi = 0; pi < 3; ++pi) {
            const int p = PP[pi], q = QQ[pi];
            const float apq = Am[p][q];
            if (fabsf(apq) < 1e-18f) continue;
            const float tau = (Am[q][q] - Am[p][p]) / (2.0f * apq);
            const float tt  = (tau >= 0.f ? 1.f : -1.f) /
                              (fabsf(tau) + sqrtf(1.f + tau * tau));
            const float c = 1.0f / sqrtf(1.f + tt * tt);
            const float s = tt * c;
            for (int k = 0; k < 3; ++k) {
                float akp = Am[k][p], akq = Am[k][q];
                Am[k][p] = c * akp - s * akq;
                Am[k][q] = s * akp + c * akq;
            }
            for (int k = 0; k < 3; ++k) {
                float apk = Am[p][k], aqk = Am[q][k];
                Am[p][k] = c * apk - s * aqk;
                Am[q][k] = s * apk + c * aqk;
            }
            for (int k = 0; k < 3; ++k) {
                float vkp = V[k][p], vkq = V[k][q];
                V[k][p] = c * vkp - s * vkq;
                V[k][q] = s * vkp + c * vkq;
            }
        }
    }

    // sort eigenvalues descending
    float ev[3] = {Am[0][0], Am[1][1], Am[2][2]};
    int idx[3] = {0, 1, 2};
    if (ev[idx[0]] < ev[idx[1]]) { int t = idx[0]; idx[0] = idx[1]; idx[1] = t; }
    if (ev[idx[0]] < ev[idx[2]]) { int t = idx[0]; idx[0] = idx[2]; idx[2] = t; }
    if (ev[idx[1]] < ev[idx[2]]) { int t = idx[1]; idx[1] = idx[2]; idx[2] = t; }

    float Vs[3][3];
    for (int i = 0; i < 3; ++i)
        for (int k = 0; k < 3; ++k)
            Vs[i][k] = V[i][idx[k]];

    // U columns = normalize(H * v_k)
    float U[3][3];
    for (int k = 0; k < 3; ++k) {
        float u[3];
        for (int i = 0; i < 3; ++i)
            u[i] = Hm[i][0] * Vs[0][k] + Hm[i][1] * Vs[1][k] + Hm[i][2] * Vs[2][k];
        float nn = sqrtf(u[0] * u[0] + u[1] * u[1] + u[2] * u[2]);
        float inv = (nn > 1e-20f) ? (1.0f / nn) : 0.f;
        for (int i = 0; i < 3; ++i) U[i][k] = u[i] * inv;
    }

    // r = V U^T ; reflection fix on V's third column ; R = Vfixed U^T
    float r0m[3][3];
    for (int i = 0; i < 3; ++i)
        for (int j = 0; j < 3; ++j)
            r0m[i][j] = Vs[i][0] * U[j][0] + Vs[i][1] * U[j][1] + Vs[i][2] * U[j][2];
    const float det =
        r0m[0][0] * (r0m[1][1] * r0m[2][2] - r0m[1][2] * r0m[2][1]) -
        r0m[0][1] * (r0m[1][0] * r0m[2][2] - r0m[1][2] * r0m[2][0]) +
        r0m[0][2] * (r0m[1][0] * r0m[2][1] - r0m[1][1] * r0m[2][0]);
    const float sgn = (det < 0.f) ? -1.f : 1.f;
    for (int i = 0; i < 3; ++i) Vs[i][2] *= sgn;

    float R[3][3];
    for (int i = 0; i < 3; ++i)
        for (int j = 0; j < 3; ++j)
            R[i][j] = Vs[i][0] * U[j][0] + Vs[i][1] * U[j][1] + Vs[i][2] * U[j][2];

    for (int i = 0; i < 3; ++i)
        for (int j = 0; j < 3; ++j)
            Rout[b * 9 + i * 3 + j] = R[i][j];
    for (int i = 0; i < 3; ++i)
        tout[b * 3 + i] =
            -(R[i][0] * sm[0] + R[i][1] * sm[1] + R[i][2] * sm[2]) + cm[i];
}

// =====================================================================
// Host entry
// =====================================================================
extern "C" void kernel_launch(void* const* d_in, const int* in_sizes, int n_in,
                              void* d_out, int out_size, void* d_ws, size_t ws_size,
                              hipStream_t stream) {
    (void)in_sizes; (void)n_in; (void)out_size; (void)ws_size;

    const float* srcE = (const float*)d_in[0];  // [B, D, N]
    const float* tgtE = (const float*)d_in[1];  // [B, D, M]
    const float* src  = (const float*)d_in[2];  // [B, 3, N]
    const float* tgt  = (const float*)d_in[3];  // [B, 3, M]

    float* out   = (float*)d_out;
    float* Rout  = out;                 // 32*9  = 288 floats
    float* tout  = out + BB * 9;        // 32*3  =  96 floats
    float* score = out + BB * 9 + BB * 3;  // [B, N, M]

    float* corr = (float*)d_ws;         // [B, 3, N] = 786432 bytes

    dim3 g1(MM / 128, NN / 64, BB);     // (16, 32, 32)
    svdhead_scores_wmma<<<g1, 256, 0, stream>>>(srcE, tgtE, score);

    svdhead_softmax_corr<<<BB * NN, 256, 0, stream>>>(score, tgt, corr);

    svdhead_kabsch<<<BB, 256, 0, stream>>>(src, corr, Rout, tout);
}
